// SparseMoE_5686536700609
// MI455X (gfx1250) — compile-verified, matrix operand order checked by reading, not run on hardware
//
#include <hip/hip_runtime.h>
#include <hip/hip_bf16.h>
#include <math.h>

typedef __bf16 bf16;
typedef __attribute__((ext_vector_type(16))) __bf16 v16bf;
typedef __attribute__((ext_vector_type(8)))  float  v8f;

#define TOKENS    4096      // B*T
#define CDIM      1024
#define NEXP      8
#define DFF       4096
#define TM        32        // rows (expert-list entries) per workgroup
#define HC        128       // DFF chunk per iteration (16 cols per wave * 8 waves)
#define HGROUPS   2         // DFF split across blockIdx.z for parallelism
#define LDH       144       // 128 + 16 pad (row stride multiple of 16B, bank-skew)
#define YROWS     (2*TOKENS + 1)
#define DUMMY_ROW (2*TOKENS)

union Frag { v16bf v; uint4 q[2]; };

// X tile in LDS: 32 rows x 1024 bf16, XOR-swizzled in 16B (8-element) blocks.
// element (row,c) lives at row*1024 + (((c>>3) ^ (row&7))<<3 | (c&7)).
__device__ __forceinline__ const uint4* xfrag(const bf16* Xl, int row, int blk) {
    return (const uint4*)(Xl + (row << 10) + ((blk ^ (row & 7)) << 3));
}

// ---------------- kernel 0: zero per-expert counters ----------------
__global__ void moe_zero(int* cnt) {
    if (threadIdx.x < NEXP) cnt[threadIdx.x] = 0;
}

// ---------------- kernel 1: router (wave per token) ----------------
__global__ __launch_bounds__(256) void moe_router(
    const float* __restrict__ x, const float* __restrict__ Wr,
    const float* __restrict__ br, int* __restrict__ cnt,
    int* __restrict__ rowsL, float* __restrict__ wtsL)
{
    const int lane = threadIdx.x & 31;
    const int tok  = blockIdx.x * 8 + (threadIdx.x >> 5);
    const float* xr = x + (size_t)tok * CDIM;

    float acc[8];
#pragma unroll
    for (int e = 0; e < 8; ++e) acc[e] = 0.0f;

    for (int c = lane; c < CDIM; c += 32) {
        const float xv = xr[c];
        const float4 w0 = *(const float4*)(Wr + c * 8);
        const float4 w1 = *(const float4*)(Wr + c * 8 + 4);
        acc[0] += xv * w0.x; acc[1] += xv * w0.y; acc[2] += xv * w0.z; acc[3] += xv * w0.w;
        acc[4] += xv * w1.x; acc[5] += xv * w1.y; acc[6] += xv * w1.z; acc[7] += xv * w1.w;
    }
#pragma unroll
    for (int off = 16; off; off >>= 1)
#pragma unroll
        for (int e = 0; e < 8; ++e) acc[e] += __shfl_xor(acc[e], off);

    if (lane == 0) {
        float s[8];
#pragma unroll
        for (int e = 0; e < 8; ++e) s[e] = acc[e] + br[e];
        // top-2, first-on-tie (matches lax.top_k); softmax monotonic in logits
        int i0 = 0;
#pragma unroll
        for (int e = 1; e < 8; ++e) if (s[e] > s[i0]) i0 = e;
        int i1 = -1; float bs = -3.4e38f;
#pragma unroll
        for (int e = 0; e < 8; ++e) if (e != i0 && s[e] > bs) { bs = s[e]; i1 = e; }
        const float e1 = expf(s[i1] - s[i0]);            // <= 1, stable
        const float w0 = 1.0f / (1.0f + e1);
        const float w1 = e1 * w0;
        int p0 = atomicAdd(&cnt[i0], 1);
        rowsL[i0 * TOKENS + p0] = tok * 2;     wtsL[i0 * TOKENS + p0] = w0;
        int p1 = atomicAdd(&cnt[i1], 1);
        rowsL[i1 * TOKENS + p1] = tok * 2 + 1; wtsL[i1 * TOKENS + p1] = w1;
    }
}

// -------- kernel 2: fp32 -> bf16 transpose (in [E][R][S] -> out [E][S][R]) --------
__global__ __launch_bounds__(256) void moe_transpose_cvt(
    const float* __restrict__ in, bf16* __restrict__ out, int R, int S)
{
    __shared__ float tile[32][33];
    const int e = blockIdx.z;
    const float* pin = in + (size_t)e * R * S;
    bf16* pout = out + (size_t)e * R * S;
    const int s0 = blockIdx.x * 32, r0 = blockIdx.y * 32;
#pragma unroll
    for (int j = 0; j < 4; ++j) {
        const int r = r0 + threadIdx.y + j * 8;
        tile[threadIdx.y + j * 8][threadIdx.x] = pin[(size_t)r * S + s0 + threadIdx.x];
    }
    __syncthreads();
#pragma unroll
    for (int j = 0; j < 4; ++j) {
        const int s = s0 + threadIdx.y + j * 8;
        pout[(size_t)s * R + r0 + threadIdx.x] = (bf16)tile[threadIdx.x][threadIdx.y + j * 8];
    }
}

// ---------------- kernel 3: fused grouped expert FFN (WMMA bf16) ----------------
// grid = (128 row-tiles, NEXP experts, HGROUPS dff-groups), 256 threads = 8 wave32s.
__global__ __launch_bounds__(256) void moe_ffn(
    const float* __restrict__ x, const bf16* __restrict__ W1t, const bf16* __restrict__ W2t,
    const float* __restrict__ b1, const float* __restrict__ b2,
    const int* __restrict__ cnt, const int* __restrict__ rowsL, const float* __restrict__ wtsL,
    float* __restrict__ y)
{
    const int e    = blockIdx.y;
    const int g    = blockIdx.z;
    const int base = blockIdx.x * TM;
    const int n    = cnt[e];
    if (base >= n) return;

    __shared__ bf16  Xl[TM * CDIM];   // 64 KB, XOR-swizzled
    __shared__ bf16  Hl[TM * LDH];    // GeLU(hidden) chunk, bf16
    __shared__ int   rT[TM];
    __shared__ float wT[TM];
    __shared__ int   tT[TM];

    const int tid = threadIdx.x;
    if (tid < TM) {
        const int idx = base + tid;
        int r = DUMMY_ROW; float w = 0.0f; int t = 0;
        if (idx < n) { r = rowsL[e * TOKENS + idx]; w = wtsL[e * TOKENS + idx]; t = r >> 1; }
        rT[tid] = r; wT[tid] = w; tT[tid] = t;
    }
    __syncthreads();

    // gather + convert X rows into LDS (coalesced along c), XOR-swizzled blocks
    for (int i = tid; i < TM * CDIM; i += 256) {
        const int row = i >> 10, c = i & (CDIM - 1);
        const float v = x[(size_t)tT[row] * CDIM + c];
        Xl[(row << 10) + ((((c >> 3) ^ (row & 7)) << 3) | (c & 7))] = (bf16)v;
    }
    __syncthreads();

    const int wave = tid >> 5;
    const int lane = tid & 31;
    const int half = lane >> 4;   // K-half per the 16-bit A/B WMMA VGPR layout
    const int lm   = lane & 15;   // M (A rows) / N (B,C cols)

    const v8f vzero = {};
    v8f out0[8], out1[8];
#pragma unroll
    for (int t = 0; t < 8; ++t) { out0[t] = vzero; out1[t] = vzero; }

    const bf16* w1e = W1t + (size_t)e * DFF * CDIM;  // [DFF][C]
    const bf16* w2e = W2t + (size_t)e * CDIM * DFF;  // [C][DFF]

    const int hcBeg = g * (DFF / HGROUPS);
    const int hcEnd = hcBeg + (DFF / HGROUPS);
    for (int hc = hcBeg; hc < hcEnd; hc += HC) {
        // ---- GEMM1: this wave's 16 hidden columns x 32 rows, K = CDIM ----
        v8f hacc0 = vzero, hacc1 = vzero;
        const bf16* brow = w1e + (size_t)(hc + wave * 16 + lm) * CDIM;
        __builtin_prefetch(brow + (size_t)HC * CDIM, 0, 1);  // next chunk's B row
#pragma unroll 4
        for (int kk = 0; kk < CDIM; kk += 32) {
            Frag b, a0, a1;
            b.q[0] = *(const uint4*)(brow + kk + half * 8);
            b.q[1] = *(const uint4*)(brow + kk + 16 + half * 8);
            const int blk = (kk >> 3) + half;
            a0.q[0] = *xfrag(Xl, lm, blk);
            a0.q[1] = *xfrag(Xl, lm, blk + 2);
            a1.q[0] = *xfrag(Xl, lm + 16, blk);
            a1.q[1] = *xfrag(Xl, lm + 16, blk + 2);
            hacc0 = __builtin_amdgcn_wmma_f32_16x16x32_bf16(
                false, a0.v, false, b.v, (short)0, hacc0, false, false);
            hacc1 = __builtin_amdgcn_wmma_f32_16x16x32_bf16(
                false, a1.v, false, b.v, (short)0, hacc1, false, false);
        }
        // bias + exact GeLU (erf form), deposit bf16 H tile (32 x HC) in LDS
        {
            const int hcol = wave * 16 + lm;
            const float bb = b1[e * DFF + hc + hcol];
#pragma unroll
            for (int r = 0; r < 8; ++r) {
                const int m = r + half * 8;
                float v0 = hacc0[r] + bb;
                v0 = 0.5f * v0 * (1.0f + erff(v0 * 0.70710678118f));
                Hl[m * LDH + hcol] = (bf16)v0;
                float v1 = hacc1[r] + bb;
                v1 = 0.5f * v1 * (1.0f + erff(v1 * 0.70710678118f));
                Hl[(m + 16) * LDH + hcol] = (bf16)v1;
            }
        }
        __syncthreads();
        // ---- GEMM2: this wave's 128 output columns x 32 rows, K = HC ----
#pragma unroll
        for (int kk = 0; kk < HC; kk += 32) {
            Frag a0, a1;
            a0.q[0] = *(const uint4*)(Hl + lm * LDH + kk + half * 8);
            a0.q[1] = *(const uint4*)(Hl + lm * LDH + kk + 16 + half * 8);
            a1.q[0] = *(const uint4*)(Hl + (lm + 16) * LDH + kk + half * 8);
            a1.q[1] = *(const uint4*)(Hl + (lm + 16) * LDH + kk + 16 + half * 8);
#pragma unroll
            for (int t = 0; t < 8; ++t) {
                Frag b;
                const bf16* b2row =
                    w2e + (size_t)(wave * 128 + t * 16 + lm) * DFF + hc + kk;
                b.q[0] = *(const uint4*)(b2row + half * 8);
                b.q[1] = *(const uint4*)(b2row + 16 + half * 8);
                out0[t] = __builtin_amdgcn_wmma_f32_16x16x32_bf16(
                    false, a0.v, false, b.v, (short)0, out0[t], false, false);
                out1[t] = __builtin_amdgcn_wmma_f32_16x16x32_bf16(
                    false, a1.v, false, b.v, (short)0, out1[t], false, false);
            }
        }
        __syncthreads();
    }

    // epilogue: y[g][row, :] = (acc (+ b2 if g==0)) * router_weight
    float* yg = y + (size_t)g * YROWS * CDIM;
#pragma unroll
    for (int t = 0; t < 8; ++t) {
        const int col = wave * 128 + t * 16 + lm;
        const float bb = (g == 0) ? b2[e * CDIM + col] : 0.0f;
#pragma unroll
        for (int r = 0; r < 8; ++r) {
            const int m = r + half * 8;
            yg[(size_t)rT[m] * CDIM + col]      = (out0[t][r] + bb) * wT[m];
            yg[(size_t)rT[m + 16] * CDIM + col] = (out1[t][r] + bb) * wT[m + 16];
        }
    }
}

// ---------------- kernel 4: combine slots & dff-groups per token ----------------
__global__ __launch_bounds__(256) void moe_combine(
    const float* __restrict__ y, float* __restrict__ out)
{
    const int i = blockIdx.x * 256 + threadIdx.x;
    const int t = i >> 10, c = i & (CDIM - 1);
    const float* y0 = y;
    const float* y1 = y + (size_t)YROWS * CDIM;
    const size_t a = (size_t)(2 * t) * CDIM + c;
    const size_t b = (size_t)(2 * t + 1) * CDIM + c;
    out[i] = (y0[a] + y0[b]) + (y1[a] + y1[b]);
}

// ---------------- host launcher ----------------
extern "C" void kernel_launch(void* const* d_in, const int* in_sizes, int n_in,
                              void* d_out, int out_size, void* d_ws, size_t ws_size,
                              hipStream_t stream)
{
    (void)in_sizes; (void)n_in; (void)out_size; (void)ws_size;
    const float* x  = (const float*)d_in[0];
    const float* Wr = (const float*)d_in[1];
    const float* br = (const float*)d_in[2];
    const float* W1 = (const float*)d_in[3];
    const float* b1 = (const float*)d_in[4];
    const float* W2 = (const float*)d_in[5];
    const float* b2 = (const float*)d_in[6];
    float* out = (float*)d_out;

    char* ws = (char*)d_ws;
    size_t off = 0;
    auto alloc = [&](size_t bytes) -> void* {
        void* p = ws + off;
        off = (off + bytes + 255) & ~(size_t)255;
        return p;
    };
    int*   cnt   = (int*)  alloc(NEXP * sizeof(int));
    int*   rowsL = (int*)  alloc((size_t)NEXP * TOKENS * sizeof(int));
    float* wtsL  = (float*)alloc((size_t)NEXP * TOKENS * sizeof(float));
    bf16*  W1t   = (bf16*) alloc((size_t)NEXP * DFF * CDIM * sizeof(bf16));
    bf16*  W2t   = (bf16*) alloc((size_t)NEXP * DFF * CDIM * sizeof(bf16));
    float* y     = (float*)alloc((size_t)HGROUPS * YROWS * CDIM * sizeof(float));

    moe_zero<<<1, 32, 0, stream>>>(cnt);
    moe_router<<<TOKENS / 8, 256, 0, stream>>>(x, Wr, br, cnt, rowsL, wtsL);
    dim3 tb(32, 8);
    // W1 [E][C][DFF] -> W1t [E][DFF][C]
    moe_transpose_cvt<<<dim3(DFF / 32, CDIM / 32, NEXP), tb, 0, stream>>>(W1, W1t, CDIM, DFF);
    // W2 [E][DFF][C] -> W2t [E][C][DFF]
    moe_transpose_cvt<<<dim3(CDIM / 32, DFF / 32, NEXP), tb, 0, stream>>>(W2, W2t, DFF, CDIM);
    moe_ffn<<<dim3(TOKENS / TM, NEXP, HGROUPS), 256, 0, stream>>>(
        x, W1t, W2t, b1, b2, cnt, rowsL, wtsL, y);
    moe_combine<<<(TOKENS * CDIM) / 256, 256, 0, stream>>>(y, out);
}
